// WeatherLSTM_90915867721773
// MI455X (gfx1250) — compile-verified, matrix operand order checked by reading, not run on hardware
//
#include <hip/hip_runtime.h>

// ---------------- problem constants ----------------
#define BB   256      // batch
#define TT   128      // encoder steps
#define II   64       // input dim
#define HH   2048     // hidden dim
#define OO   24       // output horizon
#define NG   8192     // 4*HH gate columns
#define KK   2112     // II + HH  (concat GEMM reduction dim), multiple of 32

// ---------------- WMMA vector types ----------------
typedef __attribute__((ext_vector_type(16))) __bf16 v16bf;
typedef __attribute__((ext_vector_type(8)))  __bf16 v8bf;
typedef __attribute__((ext_vector_type(8)))  float  v8f;

static __device__ __forceinline__ unsigned short f32_to_bf16(float f) {
    union { float f; unsigned u; } v; v.f = f;
    unsigned u = v.u;
    u += 0x7FFFu + ((u >> 16) & 1u);   // round-to-nearest-even
    return (unsigned short)(u >> 16);
}

// Load one 16-bit WMMA fragment for this lane.
// ISA layout (16-bit A 16x32 / B 32x16): lane holds K = ko..ko+7 (elems 0..7)
// and K = ko+16..ko+23 (elems 8..15), ko = (lane>>4)*8, row/col = lane&15.
static __device__ __forceinline__ v16bf load_frag(const unsigned short* __restrict__ p) {
    v8bf lo = *(const v8bf*)(p);
    v8bf hi = *(const v8bf*)(p + 16);
    return __builtin_shufflevector(lo, hi, 0,1,2,3,4,5,6,7,8,9,10,11,12,13,14,15);
}

// ---------------- one-time weight prep ----------------
// WcombT[n][k] = bf16( k < II ? Wx[k][n] : Wh[k-II][n] ), row stride KK.
// bias[n] = bx[n] + bh[n]
__global__ __launch_bounds__(256) void prep_weights(
        const float* __restrict__ Wx, const float* __restrict__ Wh,
        const float* __restrict__ bx, const float* __restrict__ bh,
        unsigned short* __restrict__ wT, float* __restrict__ bias) {
    int idx = blockIdx.x * 256 + threadIdx.x;            // over NG*KK
    int n = idx / KK;
    int k = idx - n * KK;
    float v = (k < II) ? Wx[(size_t)k * NG + n] : Wh[(size_t)(k - II) * NG + n];
    wT[(size_t)n * KK + k] = f32_to_bf16(v);
    if (idx < NG) bias[idx] = bx[idx] + bh[idx];
}

// zero c, h (fp32) and the h-part of the bf16 concat activation buffer
__global__ __launch_bounds__(256) void init_states(
        float* __restrict__ c, float* __restrict__ h, unsigned short* __restrict__ xh) {
    int idx = blockIdx.x * 256 + threadIdx.x;            // over BB*HH
    int b = idx >> 11;
    int j = idx & (HH - 1);
    c[idx] = 0.0f;
    h[idx] = 0.0f;
    xh[(size_t)b * KK + II + j] = 0;
}

// encoder input injection: xh[b][0:II] = bf16(trains[b][t][:])
__global__ __launch_bounds__(256) void set_x_enc(
        const float* __restrict__ trains, unsigned short* __restrict__ xh, int t) {
    int idx = blockIdx.x * 256 + threadIdx.x;            // over BB*II
    int b = idx >> 6;
    int i = idx & (II - 1);
    xh[(size_t)b * KK + i] = f32_to_bf16(trains[((size_t)b * TT + t) * II + i]);
}

// decoder input injection: xh[b][0:II] = bf16(dec[b][d]*Win[i] + b_in[i])
__global__ __launch_bounds__(256) void set_x_dec(
        const float* __restrict__ dec, const float* __restrict__ Win,
        const float* __restrict__ b_in, unsigned short* __restrict__ xh, int d) {
    int idx = blockIdx.x * 256 + threadIdx.x;            // over BB*II
    int b = idx >> 6;
    int i = idx & (II - 1);
    float x = dec[(size_t)b * (OO - 1) + d] * Win[i] + b_in[i];
    xh[(size_t)b * KK + i] = f32_to_bf16(x);
}

// ---------------- core recurrent GEMM: g[B][NG] = xh[B][KK] @ WcombT^T ----------------
// Wave tile 32(M) x 64(N): 2 A frags, 4 B frags, 8 v_wmma_f32_16x16x32_bf16 per k-step.
// Block = 8 waves arranged 4M x 2N -> 128 x 128 block tile (square tile minimizes
// combined A+B L2 traffic: 138 MB/step vs 173 MB for 64x256).
// Grid = (NG/128, BB/128) = (64, 2) = 128 blocks.
__global__ __launch_bounds__(256) void gemm_gates(
        const unsigned short* __restrict__ xh,
        const unsigned short* __restrict__ wT,
        float* __restrict__ g) {
    const int lane  = threadIdx.x & 31;
    const int wave  = threadIdx.x >> 5;
    const int wm    = wave & 3;            // 0..3  (M)
    const int wn    = wave >> 2;           // 0..1  (N)
    const int mbase = blockIdx.y * 128 + wm * 32;
    const int nbase = blockIdx.x * 128 + wn * 64;
    const int lrow  = lane & 15;
    const int khalf = (lane >> 4) * 8;

    const unsigned short* a0p = xh + (size_t)(mbase + lrow) * KK + khalf;
    const unsigned short* a1p = a0p + (size_t)16 * KK;
    const unsigned short* bp0 = wT + (size_t)(nbase +  0 + lrow) * KK + khalf;
    const unsigned short* bp1 = wT + (size_t)(nbase + 16 + lrow) * KK + khalf;
    const unsigned short* bp2 = wT + (size_t)(nbase + 32 + lrow) * KK + khalf;
    const unsigned short* bp3 = wT + (size_t)(nbase + 48 + lrow) * KK + khalf;

    v8f acc[2][4];
#pragma unroll
    for (int m = 0; m < 2; ++m)
#pragma unroll
        for (int n = 0; n < 4; ++n) acc[m][n] = (v8f){0.f,0.f,0.f,0.f,0.f,0.f,0.f,0.f};

    for (int k = 0; k < KK; k += 32) {
        v16bf a0 = load_frag(a0p + k);
        v16bf a1 = load_frag(a1p + k);
        v16bf b0 = load_frag(bp0 + k);
        v16bf b1 = load_frag(bp1 + k);
        v16bf b2 = load_frag(bp2 + k);
        v16bf b3 = load_frag(bp3 + k);
        acc[0][0] = __builtin_amdgcn_wmma_f32_16x16x32_bf16(false, a0, false, b0, (short)0, acc[0][0], false, false);
        acc[0][1] = __builtin_amdgcn_wmma_f32_16x16x32_bf16(false, a0, false, b1, (short)0, acc[0][1], false, false);
        acc[0][2] = __builtin_amdgcn_wmma_f32_16x16x32_bf16(false, a0, false, b2, (short)0, acc[0][2], false, false);
        acc[0][3] = __builtin_amdgcn_wmma_f32_16x16x32_bf16(false, a0, false, b3, (short)0, acc[0][3], false, false);
        acc[1][0] = __builtin_amdgcn_wmma_f32_16x16x32_bf16(false, a1, false, b0, (short)0, acc[1][0], false, false);
        acc[1][1] = __builtin_amdgcn_wmma_f32_16x16x32_bf16(false, a1, false, b1, (short)0, acc[1][1], false, false);
        acc[1][2] = __builtin_amdgcn_wmma_f32_16x16x32_bf16(false, a1, false, b2, (short)0, acc[1][2], false, false);
        acc[1][3] = __builtin_amdgcn_wmma_f32_16x16x32_bf16(false, a1, false, b3, (short)0, acc[1][3], false, false);
    }

    // D layout: row = v + 8*(lane>>4), col = lane&15
    const int srow = (lane >> 4) * 8;
    const int scol = lane & 15;
#pragma unroll
    for (int m = 0; m < 2; ++m)
#pragma unroll
        for (int n = 0; n < 4; ++n)
#pragma unroll
            for (int v = 0; v < 8; ++v)
                g[(size_t)(mbase + m * 16 + srow + v) * NG + nbase + n * 16 + scol] = acc[m][n][v];
}

// ---------------- LSTM cell nonlinearity ----------------
__global__ __launch_bounds__(256) void lstm_cell(
        const float* __restrict__ g, const float* __restrict__ bias,
        float* __restrict__ c, float* __restrict__ h,
        unsigned short* __restrict__ xh) {
    int idx = blockIdx.x * 256 + threadIdx.x;            // over BB*HH
    int b = idx >> 11;
    int j = idx & (HH - 1);
    const float* gb = g + (size_t)b * NG;
    float ig = gb[j]          + bias[j];
    float fg = gb[HH + j]     + bias[HH + j];
    float og = gb[2 * HH + j] + bias[2 * HH + j];
    float cg = gb[3 * HH + j] + bias[3 * HH + j];
    float si = 1.0f / (1.0f + __expf(-ig));
    float sf = 1.0f / (1.0f + __expf(-fg));
    float so = 1.0f / (1.0f + __expf(-og));
    float cn = sf * c[idx] + si * tanhf(cg);
    c[idx] = cn;
    float hn = so * tanhf(cn);
    h[idx] = hn;
    xh[(size_t)b * KK + II + j] = f32_to_bf16(hn);
}

// ---------------- output projection: y[b][col] = h[b] . Wout + bout ----------------
__global__ __launch_bounds__(256) void out_proj(
        const float* __restrict__ h, const float* __restrict__ Wout,
        const float* __restrict__ bout, float* __restrict__ y, int col) {
    __shared__ float red[256];
    int b = blockIdx.x;
    int tid = threadIdx.x;
    float s = 0.0f;
    for (int j = tid; j < HH; j += 256) s += h[(size_t)b * HH + j] * Wout[j];
    red[tid] = s;
    __syncthreads();
    for (int off = 128; off > 0; off >>= 1) {
        if (tid < off) red[tid] += red[tid + off];
        __syncthreads();
    }
    if (tid == 0) y[(size_t)b * OO + col] = red[0] + bout[0];
}

// ---------------- host orchestration ----------------
extern "C" void kernel_launch(void* const* d_in, const int* in_sizes, int n_in,
                              void* d_out, int out_size, void* d_ws, size_t ws_size,
                              hipStream_t stream) {
    (void)in_sizes; (void)n_in; (void)out_size; (void)ws_size;
    const float* trains = (const float*)d_in[0];
    const float* dec    = (const float*)d_in[1];
    const float* Wx     = (const float*)d_in[2];
    const float* bx     = (const float*)d_in[3];
    const float* Wh     = (const float*)d_in[4];
    const float* bh     = (const float*)d_in[5];
    const float* Win    = (const float*)d_in[6];
    const float* b_in   = (const float*)d_in[7];
    const float* Wout   = (const float*)d_in[8];
    const float* bout   = (const float*)d_in[9];
    float* y = (float*)d_out;

    // workspace layout (all offsets 256B aligned)
    char* ws = (char*)d_ws;
    unsigned short* wT   = (unsigned short*)(ws);                       // NG*KK*2   = 34,603,008
    unsigned short* xh   = (unsigned short*)(ws + 34603008);            // BB*KK*2   =  1,081,344
    float*          cst  = (float*)(ws + 35684352);                     // BB*HH*4   =  2,097,152
    float*          hst  = (float*)(ws + 37781504);                     // BB*HH*4   =  2,097,152
    float*          gbuf = (float*)(ws + 39878656);                     // BB*NG*4   =  8,388,608
    float*          bias = (float*)(ws + 48267264);                     // NG*4      =     32,768

    dim3 blk(256);
    dim3 gridPrep((NG * KK) / 256);
    dim3 gridInit((BB * HH) / 256);
    dim3 gridX((BB * II) / 256);
    dim3 gridGemm(NG / 128, BB / 128);
    dim3 gridCell((BB * HH) / 256);
    dim3 gridOut(BB);

    prep_weights<<<gridPrep, blk, 0, stream>>>(Wx, Wh, bx, bh, wT, bias);
    init_states<<<gridInit, blk, 0, stream>>>(cst, hst, xh);
    set_x_enc<<<gridX, blk, 0, stream>>>(trains, xh, 0);

    // encoder
    for (int t = 0; t < TT; ++t) {
        gemm_gates<<<gridGemm, blk, 0, stream>>>(xh, wT, gbuf);
        lstm_cell<<<gridCell, blk, 0, stream>>>(gbuf, bias, cst, hst, xh);
        if (t + 1 < TT) set_x_enc<<<gridX, blk, 0, stream>>>(trains, xh, t + 1);
    }
    out_proj<<<gridOut, blk, 0, stream>>>(hst, Wout, bout, y, 0);

    // decoder
    for (int d = 0; d < OO - 1; ++d) {
        set_x_dec<<<gridX, blk, 0, stream>>>(dec, Win, b_in, xh, d);
        gemm_gates<<<gridGemm, blk, 0, stream>>>(xh, wT, gbuf);
        lstm_cell<<<gridCell, blk, 0, stream>>>(gbuf, bias, cst, hst, xh);
        out_proj<<<gridOut, blk, 0, stream>>>(hst, Wout, bout, y, d + 1);
    }
}
